// QuantizedBlock_63136019251235
// MI455X (gfx1250) — 8-bit matrix operand order repaired and checked by reading, compiled, not run on hardware
//
#include <hip/hip_runtime.h>
#include <cstdint>

typedef int v8i __attribute__((ext_vector_type(8)));

#if __has_builtin(__builtin_amdgcn_global_load_async_to_lds_b128)
#define HAS_ASYNC_LDS 1
typedef int v4i_b __attribute__((vector_size(16)));
typedef __attribute__((address_space(1))) v4i_b g_v4i;
typedef __attribute__((address_space(3))) v4i_b l_v4i;
#else
#define HAS_ASYNC_LDS 0
#endif

__device__ __forceinline__ void wait_async0()
{
#if HAS_ASYNC_LDS
#if __has_builtin(__builtin_amdgcn_s_wait_asynccnt)
    __builtin_amdgcn_s_wait_asynccnt(0);
#else
    asm volatile("s_wait_asynccnt 0x0" ::: "memory");
#endif
#endif
}

__device__ __forceinline__ int sat8i(int v) { return v < -128 ? -128 : (v > 127 ? 127 : v); }
__device__ __forceinline__ int q8rn(float v) { return sat8i(__float2int_rn(v)); }
__device__ __forceinline__ float qz4(float v) { return (float)q8rn(v * 16.0f) * 0.0625f; }
__device__ __forceinline__ float qz7(float v) { return (float)q8rn(v * 128.0f) * 0.0078125f; }

#define KSTEP 64
#define LP 80

template <int ROWS, int NT>
__device__ __forceinline__ void stage_tile(const int8_t* __restrict__ gb, int ld, int k0,
                                           int8_t* sm, int tid)
{
    constexpr int U = (ROWS * 4) / NT;
#if HAS_ASYNC_LDS
#pragma unroll
    for (int u = 0; u < U; ++u) {
        const int c = tid + u * NT, row = c >> 2, off = (c & 3) * 16;
        __builtin_amdgcn_global_load_async_to_lds_b128(
            (g_v4i*)(gb + (long long)row * ld + k0 + off),
            (l_v4i*)(sm + row * LP + off), 0, 0);
    }
#else
    uint4 t[U];
#pragma unroll
    for (int u = 0; u < U; ++u) {
        const int c = tid + u * NT, row = c >> 2, off = (c & 3) * 16;
        t[u] = *(const uint4*)(gb + (long long)row * ld + k0 + off);
    }
#pragma unroll
    for (int u = 0; u < U; ++u) {
        const int c = tid + u * NT, row = c >> 2, off = (c & 3) * 16;
        *(uint4*)&sm[row * LP + off] = t[u];
    }
#endif
}

template <int WMW, int WNW, int EPI>
__global__ __launch_bounds__(32 * WMW * WNW)
void gemm_i8_wmma(const int8_t* __restrict__ A, int lda, long long sAb, long long sAh,
                  const int8_t* __restrict__ W, int ldb, long long sBb, long long sBh,
                  int8_t* __restrict__ Oq, float* __restrict__ Of, int ldc,
                  long long sCb, long long sCh,
                  const float* __restrict__ bias, const float* __restrict__ resid,
                  int K)
{
    constexpr int BM = WMW * 32, BN = WNW * 32, NT = 32 * WMW * WNW;
    __shared__ __align__(16) int8_t smA[2][BM * LP];
    __shared__ __align__(16) int8_t smB[2][BN * LP];

    const int tid  = threadIdx.x;
    const int wave = tid >> 5, lane = tid & 31;
    const int wrow = wave / WNW, wcol = wave % WNW;
    const int hl   = lane >> 4, l16 = lane & 15;

    const int bb = (int)blockIdx.z >> 4, bh = (int)blockIdx.z & 15;
    const int8_t* Ab = A + sAb * bb + sAh * bh + (long long)blockIdx.y * BM * lda;
    const int8_t* Bb = W + sBb * bb + sBh * bh + (long long)blockIdx.x * BN * ldb;
    const long long cb = sCb * bb + sCh * bh;

    const v8i vzero = {0, 0, 0, 0, 0, 0, 0, 0};
    v8i acc[2][2];
#pragma unroll
    for (int i = 0; i < 2; ++i)
#pragma unroll
        for (int j = 0; j < 2; ++j) acc[i][j] = vzero;

    stage_tile<BM, NT>(Ab, lda, 0, smA[0], tid);
    stage_tile<BN, NT>(Bb, ldb, 0, smB[0], tid);

    int cur = 0;
    for (int k0 = 0; k0 < K; k0 += KSTEP) {
        wait_async0();
        __syncthreads();

        const int kn = k0 + KSTEP;
        if (kn < K) {
            stage_tile<BM, NT>(Ab, lda, kn, smA[cur ^ 1], tid);
            stage_tile<BN, NT>(Bb, ldb, kn, smB[cur ^ 1], tid);
        }

        v8i afr[2], bfr[2];
#pragma unroll
        for (int i = 0; i < 2; ++i) {
            const int8_t* ap = smA[cur] + (wrow * 32 + i * 16 + l16) * LP + 16 * hl;
#pragma unroll
            for (int p = 0; p < 4; ++p) {
                const int* lp = (const int*)(ap + 32 * (p >> 1) + 8 * (p & 1));
                afr[i][2 * p]     = lp[0];
                afr[i][2 * p + 1] = lp[1];
            }
        }
#pragma unroll
        for (int j = 0; j < 2; ++j) {
            const int8_t* bp = smB[cur] + (wcol * 32 + j * 16 + l16) * LP + 16 * hl;
#pragma unroll
            for (int q = 0; q < 2; ++q) {
                const int4 t = *(const int4*)(bp + 32 * q);
                bfr[j][4 * q]     = t.x;
                bfr[j][4 * q + 1] = t.y;
                bfr[j][4 * q + 2] = t.z;
                bfr[j][4 * q + 3] = t.w;
            }
        }

#pragma unroll
        for (int i = 0; i < 2; ++i)
#pragma unroll
            for (int j = 0; j < 2; ++j)
                acc[i][j] = __builtin_amdgcn_wmma_i32_16x16x64_iu8(
                    true, afr[i], true, bfr[j], acc[i][j], false, false);

        cur ^= 1;
    }

    const int gm0 = (int)blockIdx.y * BM + wrow * 32;
    const int gn0 = (int)blockIdx.x * BN + wcol * 32;
#pragma unroll
    for (int i = 0; i < 2; ++i) {
#pragma unroll
        for (int j = 0; j < 2; ++j) {
            const int col = gn0 + j * 16 + l16;
#pragma unroll
            for (int r = 0; r < 8; ++r) {
                const int row = gm0 + i * 16 + hl * 8 + r;
                const long long o = cb + (long long)row * ldc + col;
                const int a = acc[i][j][r];
                if (EPI == 0) {
                    Oq[o] = (int8_t)q8rn((float)a * 0.0078125f);
                } else {
                    const float v = (float)a * (1.0f / 2048.0f) + qz7(bias[col]);
                    if (EPI == 1) {
                        Oq[o] = (int8_t)q8rn(fmaxf(v, 0.0f) * 16.0f);
                    } else {
                        Of[o] = qz4(resid[(long long)row * ldc + col]) + qz4(v);
                    }
                }
            }
        }
    }
}

__global__ void quant_w8(const float* __restrict__ w, int8_t* __restrict__ o, long long n)
{
    const long long i = (long long)blockIdx.x * blockDim.x + threadIdx.x;
    if (i < n) o[i] = (int8_t)q8rn(w[i] * 128.0f);
}

__global__ void affine_quant(const float* __restrict__ x, const float* __restrict__ aw,
                             const float* __restrict__ ab, int8_t* __restrict__ o,
                             long long n)
{
    const long long i = (long long)blockIdx.x * blockDim.x + threadIdx.x;
    if (i >= n) return;
    const float wq = qz4(aw[0]);
    const int   bi = q8rn(ab[0] * 16.0f);
    const int   ti = q8rn(qz4(x[i]) * wq * 16.0f);
    o[i] = (int8_t)sat8i(ti + bi);
}

__global__ void transpose_v(const int8_t* __restrict__ kv, int8_t* __restrict__ vT)
{
    const long long i = (long long)blockIdx.x * blockDim.x + threadIdx.x;
    const int m  = (int)(i & 1023);
    const int d  = (int)((i >> 10) & 63);
    const int bh = (int)(i >> 16);
    const int b = bh >> 4, h = bh & 15;
    vT[i] = kv[((long long)(b * 1024 + m)) * 2048 + 1024 + h * 64 + d];
}

__global__ __launch_bounds__(256) void softmax_rowq(int8_t* __restrict__ buf)
{
    const long long row = blockIdx.x;
    int8_t* p = buf + row * 1024;
    __shared__ float red[256];
    const int t = threadIdx.x;

    const int w = *(const int*)(p + 4 * t);
    const int l0 = (int)(int8_t)(w), l1 = (int)(int8_t)(w >> 8),
              l2 = (int)(int8_t)(w >> 16), l3 = (int)(int8_t)(w >> 24);

    float lm = fmaxf(fmaxf((float)l0, (float)l1), fmaxf((float)l2, (float)l3));
    red[t] = lm;
    __syncthreads();
    for (int s = 128; s > 0; s >>= 1) {
        if (t < s) red[t] = fmaxf(red[t], red[t + s]);
        __syncthreads();
    }
    const float mx = red[0];
    __syncthreads();

    const float e0 = __expf(((float)l0 - mx) * 0.0625f);
    const float e1 = __expf(((float)l1 - mx) * 0.0625f);
    const float e2 = __expf(((float)l2 - mx) * 0.0625f);
    const float e3 = __expf(((float)l3 - mx) * 0.0625f);
    red[t] = e0 + e1 + e2 + e3;
    __syncthreads();
    for (int s = 128; s > 0; s >>= 1) {
        if (t < s) red[t] += red[t + s];
        __syncthreads();
    }
    const float inv = 128.0f / red[0];

    const int a0 = q8rn(e0 * inv), a1 = q8rn(e1 * inv),
              a2 = q8rn(e2 * inv), a3 = q8rn(e3 * inv);
    *(int*)(p + 4 * t) =
        (a0 & 0xff) | ((a1 & 0xff) << 8) | ((a2 & 0xff) << 16) | ((a3 & 0xff) << 24);
}

extern "C" void kernel_launch(void* const* d_in, const int* in_sizes, int n_in,
                              void* d_out, int out_size, void* d_ws, size_t ws_size,
                              hipStream_t stream)
{
    (void)in_sizes; (void)n_in; (void)out_size; (void)ws_size;

    const float* x      = (const float*)d_in[0];
    const float* q_w    = (const float*)d_in[1];
    const float* kv_w   = (const float*)d_in[2];
    const float* proj_w = (const float*)d_in[3];
    const float* proj_b = (const float*)d_in[4];
    const float* fc1_w  = (const float*)d_in[5];
    const float* fc1_b  = (const float*)d_in[6];
    const float* fc2_w  = (const float*)d_in[7];
    const float* fc2_b  = (const float*)d_in[8];
    const float* a1w    = (const float*)d_in[9];
    const float* a1b    = (const float*)d_in[10];
    const float* a2w    = (const float*)d_in[11];
    const float* a2b    = (const float*)d_in[12];

    constexpr long long MB = 1ll << 20;
    uint8_t* ws = (uint8_t*)d_ws;
    int8_t* hq    = (int8_t*)(ws + 0 * MB);
    int8_t* wq_q  = (int8_t*)(ws + 4 * MB);
    int8_t* wq_kv = (int8_t*)(ws + 5 * MB);
    int8_t* wq_pr = (int8_t*)(ws + 7 * MB);
    int8_t* wq_f1 = (int8_t*)(ws + 8 * MB);
    int8_t* wq_f2 = (int8_t*)(ws + 12 * MB);
    int8_t* q_i8  = (int8_t*)(ws + 16 * MB);
    int8_t* kv_i8 = (int8_t*)(ws + 20 * MB);
    int8_t* attn  = (int8_t*)(ws + 28 * MB);
    int8_t* vT    = (int8_t*)(ws + 92 * MB);
    int8_t* z_i8  = (int8_t*)(ws + 96 * MB);
    float*  x2    = (float*)(ws + 100 * MB);
    int8_t* h2q   = (int8_t*)(ws + 116 * MB);
    int8_t* m_i8  = (int8_t*)(ws + 120 * MB);

    const int C = 1024, HID = 4096, BN_ = 4096;

    quant_w8<<<(C * C) / 256, 256, 0, stream>>>(q_w, wq_q, (long long)C * C);
    quant_w8<<<(2 * C * C) / 256, 256, 0, stream>>>(kv_w, wq_kv, 2ll * C * C);
    quant_w8<<<(C * C) / 256, 256, 0, stream>>>(proj_w, wq_pr, (long long)C * C);
    quant_w8<<<(HID * C) / 256, 256, 0, stream>>>(fc1_w, wq_f1, (long long)HID * C);
    quant_w8<<<(HID * C) / 256, 256, 0, stream>>>(fc2_w, wq_f2, (long long)HID * C);
    affine_quant<<<(BN_ * C) / 256, 256, 0, stream>>>(x, a1w, a1b, hq, (long long)BN_ * C);

    gemm_i8_wmma<2, 4, 0><<<dim3(C / 128, BN_ / 64, 1), 256, 0, stream>>>(
        hq, C, 0, 0, wq_q, C, 0, 0, q_i8, nullptr, C, 0, 0, nullptr, nullptr, C);
    gemm_i8_wmma<2, 4, 0><<<dim3(2 * C / 128, BN_ / 64, 1), 256, 0, stream>>>(
        hq, C, 0, 0, wq_kv, C, 0, 0, kv_i8, nullptr, 2 * C, 0, 0, nullptr, nullptr, C);
    transpose_v<<<(64 * 64 * 1024) / 256, 256, 0, stream>>>(kv_i8, vT);
    gemm_i8_wmma<2, 4, 0><<<dim3(1024 / 128, 1024 / 64, 64), 256, 0, stream>>>(
        q_i8, C, 1048576ll, 64ll,
        kv_i8, 2 * C, 2097152ll, 64ll,
        attn, nullptr, 1024, 16777216ll, 1048576ll, nullptr, nullptr, 64);
    softmax_rowq<<<64 * 1024, 256, 0, stream>>>(attn);
    gemm_i8_wmma<4, 2, 0><<<dim3(1, 1024 / 128, 64), 256, 0, stream>>>(
        attn, 1024, 16777216ll, 1048576ll,
        vT, 1024, 1048576ll, 65536ll,
        z_i8, nullptr, C, 1048576ll, 64ll, nullptr, nullptr, 1024);
    gemm_i8_wmma<2, 4, 2><<<dim3(C / 128, BN_ / 64, 1), 256, 0, stream>>>(
        z_i8, C, 0, 0, wq_pr, C, 0, 0, nullptr, x2, C, 0, 0, proj_b, x, C);
    affine_quant<<<(BN_ * C) / 256, 256, 0, stream>>>(x2, a2w, a2b, h2q, (long long)BN_ * C);
    gemm_i8_wmma<2, 4, 1><<<dim3(HID / 128, BN_ / 64, 1), 256, 0, stream>>>(
        h2q, C, 0, 0, wq_f1, C, 0, 0, m_i8, nullptr, HID, 0, 0, fc1_b, nullptr, C);
    gemm_i8_wmma<2, 4, 2><<<dim3(C / 128, BN_ / 64, 1), 256, 0, stream>>>(
        m_i8, HID, 0, 0, wq_f2, HID, 0, 0, nullptr, (float*)d_out, C, 0, 0, fc2_b, x2, HID);
}
